// SubClusteringNet_68642167325110
// MI455X (gfx1250) — compile-verified, matrix operand order checked by reading, not run on hardware
//
#include <hip/hip_runtime.h>

// Problem constants (from reference): N=131072, K=16, IN=256, OUT=128
#define N_TOK   131072
#define KEXP    16
#define IN_DIM  256
#define OUT_DIM 128

#define ROWS_PER_BLK 128
// padded total <= N + 16*127 = 133104 -> 1040 blocks of 128 rows
#define MOE_BLOCKS   1040
#define IDX_CAP      (MOE_BLOCKS * ROWS_PER_BLK)

typedef __bf16 bf16_t;
typedef __attribute__((ext_vector_type(16))) bf16_t v16bf;
typedef __attribute__((ext_vector_type(8)))  float  v8f;

// ---------------- bucketing kernels ----------------

__global__ __launch_bounds__(256) void init_ws(int* counts, int* cur, int* idxb) {
    int i = blockIdx.x * blockDim.x + threadIdx.x;
    if (i < KEXP) { counts[i] = 0; cur[i] = 0; }
    if (i < IDX_CAP) idxb[i] = -1;
}

__global__ __launch_bounds__(256) void count_k(const int* __restrict__ z, int* counts) {
    int i = blockIdx.x * blockDim.x + threadIdx.x;
    if (i < N_TOK) atomicAdd(&counts[z[i]], 1);
}

__global__ void offsets_k(const int* __restrict__ counts, int* off) {
    if (blockIdx.x == 0 && threadIdx.x == 0) {
        int acc = 0;
        for (int e = 0; e < KEXP; ++e) {
            off[e] = acc;
            acc += (counts[e] + ROWS_PER_BLK - 1) & ~(ROWS_PER_BLK - 1);
        }
        off[KEXP] = acc;
    }
}

__global__ __launch_bounds__(256) void scatter_k(const int* __restrict__ z,
                                                 const int* __restrict__ off,
                                                 int* cur, int* idxb) {
    int i = blockIdx.x * blockDim.x + threadIdx.x;
    if (i < N_TOK) {
        int e = z[i];
        int p = off[e] + atomicAdd(&cur[e], 1);
        idxb[p] = i;
    }
}

// ---------------- main MoE WMMA kernel ----------------
// Block = 256 threads = 8 wave32s; each wave owns one 16-row tile of a single
// expert's token segment. W1[e] staged in LDS pre-swizzled as bf16 B-fragments.

__global__ __launch_bounds__(256) void moe_k(const float* __restrict__ x,
                                             const int*   __restrict__ idxb,
                                             const int*   __restrict__ off,
                                             const float* __restrict__ W1,
                                             const float* __restrict__ b1,
                                             const float* __restrict__ W2,
                                             const float* __restrict__ b2,
                                             float* __restrict__ out) {
    __shared__ __attribute__((aligned(32))) unsigned char lds[64 * 1024];

    const int base  = blockIdx.x * ROWS_PER_BLK;
    const int total = off[KEXP];
    if (base >= total) return;

    // which expert owns this 128-aligned segment chunk
    int e = 0;
#pragma unroll
    for (int i = 1; i < KEXP; ++i) e = (base >= off[i]) ? i : e;

    const int tid    = threadIdx.x;
    const int lane   = tid & 31;
    const int wv     = tid >> 5;
    const int hi     = lane >> 4;   // k-half selector within fragment
    const int lanelo = lane & 15;

    // ---- stage W1[e] (256x128 f32) into LDS as bf16 WMMA B fragments ----
    // fragment (kstep, ntile) : 1KB = 32 lanes x 32B. Lane = (k-half<<4)|n.
    // 16-bit slot s = ((kk>>4)&1)*8 + (kk&7)  (matches 16-bit 32x16 layout).
    const float* w1 = W1 + (size_t)e * (IN_DIM * OUT_DIM);
    for (int i = tid; i < IN_DIM * OUT_DIM; i += 256) {
        int k = i >> 7;          // 0..255  (row of W1 = K dim)
        int n = i & 127;         // 0..127  (col of W1 = N dim)
        bf16_t v = (bf16_t)w1[i];
        int kstep = k >> 5, kk = k & 31;
        int ntile = n >> 4, nn = n & 15;
        int ln   = (((kk >> 3) & 1) << 4) | nn;
        int slot = (((kk >> 4) & 1) << 3) | (kk & 7);
        *(bf16_t*)(lds + (((kstep << 3) | ntile) << 10) + (ln << 5) + (slot << 1)) = v;
    }
    __syncthreads();

    // ---- per-row token gather ----
    const int rowPos = base + wv * 16 + lanelo;
    const int tok    = idxb[rowPos];           // -1 => padding row (discarded)
    const int tclamp = tok < 0 ? 0 : tok;      // clamp address; garbage row OK
    const float* xrow = x + (size_t)tclamp * IN_DIM;

    v8f zero = {0.f, 0.f, 0.f, 0.f, 0.f, 0.f, 0.f, 0.f};
    v8f acc[8];
#pragma unroll
    for (int i = 0; i < 8; ++i) acc[i] = zero;

    // ---- h = x @ W1 : 8 k-steps of 32, 8 n-tiles of 16 ----
#pragma unroll
    for (int ks = 0; ks < 8; ++ks) {
        const int kb = ks * 32 + hi * 8;       // A frag: lane holds K {kb..kb+7, kb+16..kb+23}
        float4 f0 = *(const float4*)(xrow + kb);
        float4 f1 = *(const float4*)(xrow + kb + 4);
        float4 f2 = *(const float4*)(xrow + kb + 16);
        float4 f3 = *(const float4*)(xrow + kb + 20);
        v16bf a;
        a[0]=(bf16_t)f0.x; a[1]=(bf16_t)f0.y; a[2] =(bf16_t)f0.z; a[3] =(bf16_t)f0.w;
        a[4]=(bf16_t)f1.x; a[5]=(bf16_t)f1.y; a[6] =(bf16_t)f1.z; a[7] =(bf16_t)f1.w;
        a[8]=(bf16_t)f2.x; a[9]=(bf16_t)f2.y; a[10]=(bf16_t)f2.z; a[11]=(bf16_t)f2.w;
        a[12]=(bf16_t)f3.x; a[13]=(bf16_t)f3.y; a[14]=(bf16_t)f3.z; a[15]=(bf16_t)f3.w;
#pragma unroll
        for (int nt = 0; nt < 8; ++nt) {
            v16bf bf = *(const v16bf*)(lds + (((ks << 3) | nt) << 10) + (lane << 5));
            acc[nt] = __builtin_amdgcn_wmma_f32_16x16x32_bf16(
                false, a, false, bf, (short)0, acc[nt], false, false);
        }
    }

    // ---- layer 2: relu(h+b1) @ W2 (128->2), per-lane partials then butterfly ----
    // C layout: value in acc[nt][v] at this lane is (M = v + 8*hi, N = nt*16 + lanelo)
    float s0[8], s1[8];
#pragma unroll
    for (int v = 0; v < 8; ++v) { s0[v] = 0.f; s1[v] = 0.f; }
#pragma unroll
    for (int nt = 0; nt < 8; ++nt) {
        int n = nt * 16 + lanelo;
        float bb  = b1[e * OUT_DIM + n];
        float w20 = W2[((size_t)e * OUT_DIM + n) * 2 + 0];
        float w21 = W2[((size_t)e * OUT_DIM + n) * 2 + 1];
#pragma unroll
        for (int v = 0; v < 8; ++v) {
            float h = acc[nt][v] + bb;
            h = h > 0.f ? h : 0.f;
            s0[v] += h * w20;
            s1[v] += h * w21;
        }
    }
#pragma unroll
    for (int v = 0; v < 8; ++v) {
#pragma unroll
        for (int m = 1; m < 16; m <<= 1) {     // reduce over N within 16-lane group
            s0[v] += __shfl_xor(s0[v], m, 32);
            s1[v] += __shfl_xor(s1[v], m, 32);
        }
    }

    const float b20 = b2[e * 2 + 0], b21 = b2[e * 2 + 1];
    __syncthreads();                            // done with B fragments; reuse LDS
    float* lgt = (float*)lds;                   // [wave][16 rows][2]
#pragma unroll
    for (int v = 0; v < 8; ++v) {
        if (lanelo == v) {                      // lanes v and v+16 cover M=v, M=v+8
            int M = v + 8 * hi;
            lgt[(wv * 16 + M) * 2 + 0] = s0[v] + b20;
            lgt[(wv * 16 + M) * 2 + 1] = s1[v] + b21;
        }
    }
    __syncthreads();

    if (lane < 16 && tok >= 0) {                // lane == its own row M
        float l0 = lgt[(wv * 16 + lane) * 2 + 0];
        float l1 = lgt[(wv * 16 + lane) * 2 + 1];
        float mx = fmaxf(l0, l1);
        float e0 = __expf(l0 - mx), e1 = __expf(l1 - mx);
        float inv = 1.f / (e0 + e1);
        out[(size_t)tok * 2 + 0] = e0 * inv;
        out[(size_t)tok * 2 + 1] = e1 * inv;
    }
}

// ---------------- launcher ----------------

extern "C" void kernel_launch(void* const* d_in, const int* in_sizes, int n_in,
                              void* d_out, int out_size, void* d_ws, size_t ws_size,
                              hipStream_t stream) {
    (void)in_sizes; (void)n_in; (void)out_size; (void)ws_size;
    const float* x  = (const float*)d_in[0];
    const int*   z  = (const int*)d_in[1];
    const float* W1 = (const float*)d_in[2];
    const float* b1 = (const float*)d_in[3];
    const float* W2 = (const float*)d_in[4];
    const float* b2 = (const float*)d_in[5];
    float* out = (float*)d_out;

    int* counts = (int*)d_ws;       // 16
    int* off    = counts + 16;      // 17
    int* cur    = counts + 48;      // 16
    int* idxb   = counts + 128;     // IDX_CAP

    init_ws  <<<(IDX_CAP + 255) / 256, 256, 0, stream>>>(counts, cur, idxb);
    count_k  <<<(N_TOK  + 255) / 256, 256, 0, stream>>>(z, counts);
    offsets_k<<<1, 1, 0, stream>>>(counts, off);
    scatter_k<<<(N_TOK  + 255) / 256, 256, 0, stream>>>(z, off, cur, idxb);
    moe_k    <<<MOE_BLOCKS, 256, 0, stream>>>(x, idxb, off, W1, b1, W2, b2, out);
}
